// MLPBlock_29549374996877
// MI455X (gfx1250) — matrix operand order repaired and checked by reading, not compiled, not run on hardware
//
#include <hip/hip_runtime.h>
#include <hip/hip_bf16.h>

// ---------------------------------------------------------------------------
// gpt-oss MoE MLP block for MI455X (gfx1250, wave32, WMMA, async-LDS)
//   B=1, S=2048, D=1024, I=2048 (mlp1 emits 2I interleaved), E=8, K=2
// Token binning per expert -> gathered bf16 WMMA GEMMs. 64-token tiles keep
// per-expert weight working set (25MB) resident in the 192MB L2; HBM traffic
// ~201MB (weights read once) => memory-bound near the 23.3TB/s roofline.
// ---------------------------------------------------------------------------

#define S_   2048
#define D_   1024
#define I_   2048
#define I2_  4096
#define E_   8
#define TM   64                 // tokens per tile
#define TILES_PER_E (S_ / TM)   // 32
#define LIMITF 7.0f
#define ALPHAF 1.702f
#define EPSF   1e-5f

typedef __bf16 bf16_t;
typedef __attribute__((ext_vector_type(16))) __bf16 v16bf;
typedef __attribute__((ext_vector_type(8)))  __bf16 v8bf;
typedef __attribute__((ext_vector_type(2)))  __bf16 v2bf;
typedef __attribute__((ext_vector_type(8)))  float  v8f;

// ---------------------------------------------------------------------------
// Kernel 0: residual init (out = x) + zero per-expert counters
// ---------------------------------------------------------------------------
__global__ __launch_bounds__(512) void init_kernel(const float* __restrict__ x,
                                                   float* __restrict__ out,
                                                   int* __restrict__ cnt, int n) {
  int i = blockIdx.x * blockDim.x + threadIdx.x;
  if (i < n) out[i] = x[i];
  if (i < E_) cnt[i] = 0;
}

// ---------------------------------------------------------------------------
// Kernel 1: per-token RMSNorm (-> bf16), gate logits, top-2 softmax routing.
// ---------------------------------------------------------------------------
__global__ __launch_bounds__(256) void route_kernel(
    const float* __restrict__ x, const float* __restrict__ norm_scale,
    const float* __restrict__ gate_w, const float* __restrict__ gate_b,
    bf16_t* __restrict__ t_ws, int* __restrict__ cnt,
    int* __restrict__ idxl, float* __restrict__ wl) {
  __shared__ float red[256];
  const int tok = blockIdx.x;
  const int tid = threadIdx.x;

  const float4 xv = ((const float4*)(x + (size_t)tok * D_))[tid];
  red[tid] = xv.x * xv.x + xv.y * xv.y + xv.z * xv.z + xv.w * xv.w;
  __syncthreads();
  for (int s = 128; s > 0; s >>= 1) {
    if (tid < s) red[tid] += red[tid + s];
    __syncthreads();
  }
  const float rs = rsqrtf(red[0] / (float)D_ + EPSF);

  const int d0 = tid * 4;
  float tv[4];
  tv[0] = xv.x * rs * norm_scale[d0 + 0];
  tv[1] = xv.y * rs * norm_scale[d0 + 1];
  tv[2] = xv.z * rs * norm_scale[d0 + 2];
  tv[3] = xv.w * rs * norm_scale[d0 + 3];
#pragma unroll
  for (int j = 0; j < 4; ++j) t_ws[(size_t)tok * D_ + d0 + j] = (bf16_t)tv[j];

  float acc[E_];
#pragma unroll
  for (int e = 0; e < E_; ++e) acc[e] = 0.f;
#pragma unroll
  for (int j = 0; j < 4; ++j) {
    const float* gw = gate_w + (size_t)(d0 + j) * E_;
#pragma unroll
    for (int e = 0; e < E_; ++e) acc[e] += tv[j] * gw[e];
  }
#pragma unroll
  for (int e = 0; e < E_; ++e) {
    float v = acc[e];
    for (int off = 16; off > 0; off >>= 1) v += __shfl_down(v, off, 32);
    acc[e] = v;
  }
  __syncthreads();
  const int wv = tid >> 5, ln = tid & 31;
  if (ln == 0)
#pragma unroll
    for (int e = 0; e < E_; ++e) red[wv * E_ + e] = acc[e];
  __syncthreads();

  if (tid == 0) {
    float lg[E_];
#pragma unroll
    for (int e = 0; e < E_; ++e) {
      float s = 0.f;
      for (int w = 0; w < 8; ++w) s += red[w * E_ + e];
      lg[e] = s + gate_b[e];
    }
    int i0 = 0;
    for (int e = 1; e < E_; ++e)
      if (lg[e] > lg[i0]) i0 = e;               // ties -> lower index
    int i1 = -1;
    for (int e = 0; e < E_; ++e)
      if (e != i0 && (i1 < 0 || lg[e] > lg[i1])) i1 = e;
    const float m  = fmaxf(lg[i0], lg[i1]);
    const float e0 = __expf(lg[i0] - m), e1 = __expf(lg[i1] - m);
    const float inv = 1.f / (e0 + e1);
    int p0 = atomicAdd(&cnt[i0], 1);
    idxl[i0 * S_ + p0] = tok;  wl[i0 * S_ + p0] = e0 * inv;
    int p1 = atomicAdd(&cnt[i1], 1);
    idxl[i1 * S_ + p1] = tok;  wl[i1 * S_ + p1] = e1 * inv;
  }
}

// ---------------------------------------------------------------------------
// WMMA fragment loads (ISA 7.12.2), bf16, K=32 step, from LDS (16B accesses)
// ---------------------------------------------------------------------------
__device__ __forceinline__ v16bf load_frag_a(const bf16_t* base, int rowstride,
                                             int row16, int kbase, int hi) {
  // A 16x32: lanes<16 row=lane, K {0..7,16..23}; lanes>=16 K {8..15,24..31}
  const bf16_t* p = base + row16 * rowstride + kbase + hi * 8;
  v8bf lo = *(const v8bf*)p;
  v8bf hh = *(const v8bf*)(p + 16);
  return __builtin_shufflevector(lo, hh, 0, 1, 2, 3, 4, 5, 6, 7, 8, 9, 10, 11,
                                 12, 13, 14, 15);
}
__device__ __forceinline__ v16bf load_frag_b(const bf16_t* base, int col,
                                             int stride, int kbase, int hi) {
  // B 32x16: lanes<16 col=lane K {0..7,16..23}; lanes>=16 K {8..15,24..31} ([col][k] layout)
  const bf16_t* p = base + col * stride + kbase + hi * 8;
  v8bf lo = *(const v8bf*)p;
  v8bf hh = *(const v8bf*)(p + 16);
  return __builtin_shufflevector(lo, hh, 0, 1, 2, 3, 4, 5, 6, 7, 8, 9, 10, 11,
                                 12, 13, 14, 15);
}
__device__ __forceinline__ void pack_store(bf16_t* dst, float a, float b) {
  v2bf p;
  p[0] = (bf16_t)a;
  p[1] = (bf16_t)b;                 // -> v_cvt_pk_bf16_f32 + ds_store_b32
  *(v2bf*)dst = p;
}
// CDNA5 async global->LDS copy (ASYNCcnt), per CDNA5 ISA
__device__ __forceinline__ void async_copy_b128(unsigned lds_byte,
                                                const void* gaddr) {
  asm volatile("global_load_async_to_lds_b128 %0, %1, off"
               :: "v"(lds_byte), "v"(gaddr)
               : "memory");
}
__device__ __forceinline__ void wait_async0() {
  asm volatile("s_wait_asynccnt 0x0" ::: "memory");
}

// ---------------------------------------------------------------------------
// Kernel 2: per-(expert, 64-token tile) fused mlp1 -> swiglu -> mlp2 -> scatter
// 512 threads = 16 waves. Dynamic LDS (285.5 KB of 320 KB WGP LDS):
//   Atile [64][1024] bf16          128 KB  (async-gathered activations)
//   Blds1 2 x [128n][136k] bf16     68 KB  (W1 slab, double-buffered, padded)
//   Blds2 [1024n][40k] bf16         80 KB  (W2 slab, padded)
//   Hs    [64][72] bf16              9 KB  (SwiGLU output, current I-chunk)
// ---------------------------------------------------------------------------
#define B1S   136                  // Blds1 row stride (elems), 16B-aligned pad
#define B1ELE (128 * B1S)          // elems per Blds1 buffer
#define B2S   40                   // Blds2 row stride
#define HSS   72                   // Hs row stride
#define SM_ATILE 0
#define SM_B1    131072
#define SM_B2    200704
#define SM_HS    282624
#define SM_TOK   291840
#define SM_WGT   292096
#define SM_TOTAL 292352

__global__ __launch_bounds__(512, 1) void moe_expert_kernel(
    const bf16_t* __restrict__ t_ws,
    const float* __restrict__ mlp1_w, const float* __restrict__ mlp1_b,
    const float* __restrict__ mlp2_w, const float* __restrict__ mlp2_b,
    const int* __restrict__ cnt, const int* __restrict__ idxl,
    const float* __restrict__ wl, float* __restrict__ out) {
  extern __shared__ __align__(128) char smem[];
  bf16_t* Atile = (bf16_t*)(smem + SM_ATILE);
  bf16_t* Blds1 = (bf16_t*)(smem + SM_B1);
  bf16_t* Blds2 = (bf16_t*)(smem + SM_B2);
  bf16_t* Hs    = (bf16_t*)(smem + SM_HS);
  int*    stok  = (int*)(smem + SM_TOK);
  float*  swt   = (float*)(smem + SM_WGT);

  const int e     = blockIdx.x / TILES_PER_E;
  const int tile  = blockIdx.x % TILES_PER_E;
  const int cnte  = cnt[e];
  const int rbase = tile * TM;
  if (rbase >= cnte) return;                       // uniform per block
  const int nrows = min(TM, cnte - rbase);

  const int tid = threadIdx.x;
  const int wave = tid >> 5, lane = tid & 31, nlane = lane & 15, hi = lane >> 4;

  const float* w1e = mlp1_w + (size_t)e * D_ * I2_;
  const float* b1e = mlp1_b + (size_t)e * I2_;
  const float* w2e = mlp2_w + (size_t)e * I_ * D_;
  const float* b2e = mlp2_b + (size_t)e * D_;

  if (tid < TM) {
    if (tid < nrows) {
      stok[tid] = idxl[e * S_ + rbase + tid];
      swt[tid]  = wl[e * S_ + rbase + tid];
    } else {
      stok[tid] = 0;                               // padded rows: weight 0
      swt[tid]  = 0.f;
    }
  }
  __syncthreads();

  // async-gather 64 token rows of bf16 activations into Atile (16B chunks)
  {
    const int row = tid >> 3, seg = tid & 7;       // 8 threads x 256B per row
    const bf16_t* g = t_ws + (size_t)stok[row] * D_ + seg * 128;
    const unsigned lds0 = SM_ATILE + (unsigned)(row * D_ + seg * 128) * 2;
#pragma unroll
    for (int j = 0; j < 16; ++j)
      async_copy_b128(lds0 + j * 16, g + j * 8);
    wait_async0();
  }

  // stage-2 accumulators: wave owns 4 M-tiles x 4 N-tiles (64 output cols)
  v8f C[4][4];
#pragma unroll
  for (int mt = 0; mt < 4; ++mt)
#pragma unroll
    for (int nt = 0; nt < 4; ++nt)
#pragma unroll
      for (int i = 0; i < 8; ++i) C[mt][nt][i] = 0.f;

  const int mta  = (wave & 1) * 2;                 // stage-1: 2 M-tiles / wave
  const int nt1  = wave >> 1;                      // stage-1 N-tile (0..7)
  const int col1 = nt1 * 16 + nlane;
  const int wcol = wave * 64;                      // stage-2 N base

  for (int chunk = 0; chunk < I_ / 64; ++chunk) {  // 32 chunks of 64 interm.
    // ============ stage 1: H = Atile @ W1[:, chunk*128 .. +128) ============
    v8f hc[2];
#pragma unroll
    for (int t2 = 0; t2 < 2; ++t2)
#pragma unroll
      for (int i = 0; i < 8; ++i) hc[t2][i] = 0.f;

    // fill slab 0 into buffer 0 (128k x 128n, packed k-pairs -> ds_store_b32)
#pragma unroll 4
    for (int rep = 0; rep < 16; ++rep) {
      const int lin = rep * 512 + tid;
      const int n = lin & 127, kp = lin >> 7;      // kp: 0..63
      const float* g = w1e + (size_t)(2 * kp) * I2_ + chunk * 128 + n;
      pack_store(Blds1 + n * B1S + 2 * kp, g[0], g[I2_]);
    }
    __syncthreads();

    for (int r = 0; r < 8; ++r) {                  // 8 k-slabs of 128 over D
      if (r + 1 < 8) {                             // fill next slab (ping-pong)
        bf16_t* buf = Blds1 + ((r + 1) & 1) * B1ELE;
#pragma unroll 4
        for (int rep = 0; rep < 16; ++rep) {
          const int lin = rep * 512 + tid;
          const int n = lin & 127, kp = lin >> 7;
          const float* g =
              w1e + (size_t)((r + 1) * 128 + 2 * kp) * I2_ + chunk * 128 + n;
          pack_store(buf + n * B1S + 2 * kp, g[0], g[I2_]);
        }
        __builtin_prefetch(w1e + (size_t)((r + 1) * 128) * I2_ +
                               (chunk + 1) * 128 + (tid & 127), 0, 1);
      }
      const bf16_t* buf = Blds1 + (r & 1) * B1ELE;
#pragma unroll
      for (int ks = 0; ks < 4; ++ks) {             // 4 WMMA k-steps per slab
        const v16bf bf = load_frag_b(buf, nt1 * 16 + nlane, B1S, ks * 32, hi);
#pragma unroll
        for (int t2 = 0; t2 < 2; ++t2) {
          const v16bf af = load_frag_a(Atile, D_, (mta + t2) * 16 + nlane,
                                       r * 128 + ks * 32, hi);
          hc[t2] = __builtin_amdgcn_wmma_f32_16x16x32_bf16(
              false, af, false, bf, (short)0, hc[t2], false, false);
        }
      }
      __syncthreads();                             // one barrier per phase
    }

    // bias + interleaved SwiGLU in C-register layout (pair via shfl_xor 1)
    {
      const float b1v = b1e[chunk * 128 + col1];
      const int iout = nt1 * 8 + (nlane >> 1);
#pragma unroll
      for (int t2 = 0; t2 < 2; ++t2)
#pragma unroll
        for (int r = 0; r < 8; ++r) {
          const float v = hc[t2][r] + b1v;
          const float pv = __shfl_xor(v, 1, 32);
          if ((nlane & 1) == 0) {                  // even col glu, odd linear
            const float g = fminf(v, LIMITF);
            const float l = fminf(fmaxf(pv, -LIMITF), LIMITF);
            const float act =
                g * (1.f / (1.f + __expf(-ALPHAF * g))) * (l + 1.f);
            Hs[((mta + t2) * 16 + r + 8 * hi) * HSS + iout] = (bf16_t)act;
          }
        }
    }

    // ============ stage 2: Y += Hs @ W2[chunk rows, :] =====================
#pragma unroll
    for (int kk = 0; kk < 2; ++kk) {               // 64 = 2 x K32 slabs
      __syncthreads();                             // Hs ready / Blds2 free
#pragma unroll 4
      for (int rep = 0; rep < 32; ++rep) {         // 32k x 1024n packed pairs
        const int lin = rep * 512 + tid;
        const int n = lin & 1023, kp = lin >> 10;  // kp: 0..15
        const float* g =
            w2e + (size_t)(chunk * 64 + kk * 32 + 2 * kp) * D_ + n;
        pack_store(Blds2 + n * B2S + 2 * kp, g[0], g[D_]);
      }
      __syncthreads();
#pragma unroll
      for (int mt = 0; mt < 4; ++mt) {
        const v16bf ah = load_frag_a(Hs, HSS, mt * 16 + nlane, kk * 32, hi);
#pragma unroll
        for (int nt = 0; nt < 4; ++nt) {
          const v16bf bh =
              load_frag_b(Blds2, wcol + nt * 16 + nlane, B2S, 0, hi);
          C[mt][nt] = __builtin_amdgcn_wmma_f32_16x16x32_bf16(
              false, ah, false, bh, (short)0, C[mt][nt], false, false);
        }
      }
    }
  }

  // epilogue: out[token] += w * (C + b2)   (atomic: 2 experts per token)
#pragma unroll
  for (int mt = 0; mt < 4; ++mt) {
#pragma unroll
    for (int nt = 0; nt < 4; ++nt) {
      const int gcol = wcol + nt * 16 + nlane;
      const float b2v = b2e[gcol];
#pragma unroll
      for (int r = 0; r < 8; ++r) {
        const int m = mt * 16 + r + 8 * hi;
        const float w = swt[m];
        if (w != 0.f)
          atomicAdd(&out[(size_t)stok[m] * D_ + gcol],
                    w * (C[mt][nt][r] + b2v));
      }
    }
  }
}

// ---------------------------------------------------------------------------
// host entry
// ---------------------------------------------------------------------------
extern "C" void kernel_launch(void* const* d_in, const int* in_sizes, int n_in,
                              void* d_out, int out_size, void* d_ws,
                              size_t ws_size, hipStream_t stream) {
  const float* x          = (const float*)d_in[0];
  const float* norm_scale = (const float*)d_in[1];
  const float* gate_w     = (const float*)d_in[2];
  const float* gate_b     = (const float*)d_in[3];
  const float* mlp1_w     = (const float*)d_in[4];
  const float* mlp1_b     = (const float*)d_in[5];
  const float* mlp2_w     = (const float*)d_in[6];
  const float* mlp2_b     = (const float*)d_in[7];
  float* out = (float*)d_out;

  char* ws = (char*)d_ws;
  bf16_t* t_ws = (bf16_t*)ws;                              // 4 MB
  int*    cnt  = (int*)(ws + (size_t)S_ * D_ * 2);
  int*    idxl = (int*)(ws + (size_t)S_ * D_ * 2 + 256);   // 64 KB
  float*  wl   = (float*)(ws + (size_t)S_ * D_ * 2 + 256 + (size_t)E_ * S_ * 4);

  (void)hipFuncSetAttribute((const void*)moe_expert_kernel,
                            hipFuncAttributeMaxDynamicSharedMemorySize,
                            SM_TOTAL);

  init_kernel<<<(S_ * D_ + 511) / 512, 512, 0, stream>>>(x, out, cnt, S_ * D_);
  route_kernel<<<S_, 256, 0, stream>>>(x, norm_scale, gate_w, gate_b, t_ws, cnt,
                                       idxl, wl);
  moe_expert_kernel<<<E_ * TILES_PER_E, 512, SM_TOTAL, stream>>>(
      t_ws, mlp1_w, mlp1_b, mlp2_w, mlp2_b, cnt, idxl, wl, out);
}
